// FractalBert_4793183502455
// MI455X (gfx1250) — matrix operand order repaired and checked by reading, not compiled, not run on hardware
//
#include <hip/hip_runtime.h>
#include <hip/hip_bf16.h>
#include <math.h>

// ---------------- model dims ----------------
#define SEQ    8192
#define HID    256
#define HEADS  8
#define HDIM   32
#define KWIN   64
#define FFN    1024
#define VOCAB  8192
#define LAYERS 4

// ---------------- WMMA types (CDNA5 wave32) ----------------
typedef __attribute__((ext_vector_type(16))) __bf16 v16bf;
typedef __attribute__((ext_vector_type(8)))  float  v8f;

union BF16Frag {
    unsigned short u[16];
    uint4          q[2];
    v16bf          v;
};

__device__ __forceinline__ unsigned short f2bf(float f) {
    // round-to-nearest-even f32 -> bf16
    unsigned int u = __float_as_uint(f);
    u += 0x7FFFu + ((u >> 16) & 1u);
    return (unsigned short)(u >> 16);
}

// ---------------- fp32 -> bf16 weight conversion ----------------
__global__ void cvt_bf16_kernel(const float* __restrict__ in,
                                unsigned short* __restrict__ out, int n) {
    int i = blockIdx.x * blockDim.x + threadIdx.x;
    if (i < n) out[i] = f2bf(in[i]);
}

// ---------------- embedding gather + LN + RoPE ----------------
// one block (256 threads) per token row
__global__ void embed_ln_rope_kernel(const int* __restrict__ x,
                                     const float* __restrict__ emb,
                                     const float* __restrict__ g,
                                     const float* __restrict__ b,
                                     const float* __restrict__ cantor,
                                     float* __restrict__ h,
                                     unsigned short* __restrict__ hb) {
    int s = blockIdx.x, t = threadIdx.x;
    __shared__ float red[HID];
    __shared__ float xs[HID];
    int tok = x[s];
    float e = emb[(size_t)tok * HID + t];
    red[t] = e;
    __syncthreads();
    for (int k = 128; k > 0; k >>= 1) { if (t < k) red[t] += red[t + k]; __syncthreads(); }
    float mu = red[0] * (1.0f / HID);
    __syncthreads();
    float d = e - mu;
    red[t] = d * d;
    __syncthreads();
    for (int k = 128; k > 0; k >>= 1) { if (t < k) red[t] += red[t + k]; __syncthreads(); }
    float var = red[0] * (1.0f / HID);
    float val = d * rsqrtf(var + 1e-5f) * g[t] + b[t];
    xs[t] = val;
    __syncthreads();
    // RoPE per head (D=32, 16 freqs)
    int il   = t & (HDIM - 1);
    int base = t - il;
    int j    = il >> 1;
    float inv = powf(1.0e6f, -(float)j / 16.0f);        // ROPE_MAX_PERIOD^(-2j/D)
    float ph  = cantor[s] * 100.0f * inv;               // ROPE_SCALE = 100
    float c = cosf(ph), sn = sinf(ph);
    float xr = xs[base + 2 * j], xi = xs[base + 2 * j + 1];
    float o = (il & 1) ? (xr * sn + xi * c) : (xr * c - xi * sn);
    h [(size_t)s * HID + t] = o;
    hb[(size_t)s * HID + t] = f2bf(o);
}

// ---------------- residual + LayerNorm ----------------
__global__ void residual_ln_kernel(const float* __restrict__ hin,
                                   const float* __restrict__ a,
                                   const float* __restrict__ g,
                                   const float* __restrict__ b,
                                   float* __restrict__ hout,
                                   unsigned short* __restrict__ hb) {
    int s = blockIdx.x, t = threadIdx.x;
    __shared__ float red[HID];
    float x = hin[(size_t)s * HID + t] + a[(size_t)s * HID + t];
    red[t] = x;
    __syncthreads();
    for (int k = 128; k > 0; k >>= 1) { if (t < k) red[t] += red[t + k]; __syncthreads(); }
    float mu = red[0] * (1.0f / HID);
    __syncthreads();
    float d = x - mu;
    red[t] = d * d;
    __syncthreads();
    for (int k = 128; k > 0; k >>= 1) { if (t < k) red[t] += red[t + k]; __syncthreads(); }
    float var = red[0] * (1.0f / HID);
    float val = d * rsqrtf(var + 1e-5f) * g[t] + b[t];
    hout[(size_t)s * HID + t] = val;
    hb [(size_t)s * HID + t] = f2bf(val);
}

// ---------------- windowed attention (fp32 VALU; ~2 GFLOP total) ----------------
// block = 64 threads, one block per (s, head)
__global__ void attn_kernel(const float* __restrict__ q,
                            const float* __restrict__ k,
                            const float* __restrict__ v,
                            const int* __restrict__ routes,
                            unsigned short* __restrict__ ctx) {
    int s = blockIdx.x, hh = blockIdx.y, t = threadIdx.x;
    __shared__ float qs[HDIM];
    __shared__ float sc[KWIN];
    __shared__ int   rt[KWIN];
    __shared__ float smax, sinv;
    if (t < HDIM) qs[t] = q[(size_t)s * HID + hh * HDIM + t];
    rt[t] = routes[(size_t)s * KWIN + t];
    __syncthreads();
    const float* kp = k + (size_t)rt[t] * HID + hh * HDIM;
    float dot = 0.f;
    #pragma unroll
    for (int d = 0; d < HDIM; ++d) dot += qs[d] * kp[d];
    sc[t] = dot * 0.17677669529663687f;   // 1/sqrt(32)
    __syncthreads();
    if (t == 0) {
        float m = sc[0];
        for (int i = 1; i < KWIN; ++i) m = fmaxf(m, sc[i]);
        smax = m;
    }
    __syncthreads();
    sc[t] = expf(sc[t] - smax);
    __syncthreads();
    if (t == 0) {
        float sum = 0.f;
        for (int i = 0; i < KWIN; ++i) sum += sc[i];
        sinv = 1.0f / sum;
    }
    __syncthreads();
    if (t < HDIM) {
        float o = 0.f;
        for (int i = 0; i < KWIN; ++i)
            o += sc[i] * v[(size_t)rt[i] * HID + hh * HDIM + t];
        ctx[(size_t)s * HID + hh * HDIM + t] = f2bf(o * sinv);
    }
}

// ---------------- bf16 WMMA GEMM, 2x2 register-tiled ----------------
// C = act(A @ W^T + bias).  A: M x K bf16 row-major, W: N x K bf16 row-major.
// block = 256 threads = 8 waves arranged 2(M) x 4(N); block tile = 64 x 128.
// Each wave owns a 32x32 output tile: 4 accumulators, 2 A-frags, 2 B-frags
// per K-step of 32 -> 4 WMMAs per 8 b128 loads (2 loads/WMMA).
// Fragment layouts per CDNA5 ISA 7.12.2 (16-bit A 16x32; B 32x16; f32 C/D 16x16).
#define WMMA_BF16(A_, B_, C_) \
    __builtin_amdgcn_wmma_f32_16x16x32_bf16(false, (A_).v, false, (B_).v, (short)0, (C_), false, false)

template <int ACT, int WF32, int WBF16>
__global__ __launch_bounds__(256) void gemm_bf16_wmma(
    const unsigned short* __restrict__ A, const unsigned short* __restrict__ W,
    const float* __restrict__ bias, float* __restrict__ Cf,
    unsigned short* __restrict__ Cb, int M, int N, int K) {
    const int lane = threadIdx.x & 31;
    const int wave = threadIdx.x >> 5;
    const int half = lane >> 4;   // 0 or 1
    const int l15  = lane & 15;
    const int tM   = blockIdx.y * 64 + (wave >> 2) * 32;
    const int tN   = blockIdx.x * 128 + (wave & 3) * 32;

    const unsigned short* Ar0 = A + (size_t)(tM + l15) * K;
    const unsigned short* Ar1 = A + (size_t)(tM + 16 + l15) * K;
    const unsigned short* Wr0 = W + (size_t)(tN + l15) * K;
    const unsigned short* Wr1 = W + (size_t)(tN + 16 + l15) * K;

    v8f acc00 = {0.f,0.f,0.f,0.f,0.f,0.f,0.f,0.f};
    v8f acc01 = acc00, acc10 = acc00, acc11 = acc00;

    for (int k0 = 0; k0 < K; k0 += 32) {
        BF16Frag a0, a1, b0, b1;
        // A lane layout: elem i -> K = k0 + (i>>3)*16 + half*8 + (i&7)
        a0.q[0] = *(const uint4*)(Ar0 + k0 + half * 8);
        a0.q[1] = *(const uint4*)(Ar0 + k0 + 16 + half * 8);
        a1.q[0] = *(const uint4*)(Ar1 + k0 + half * 8);
        a1.q[1] = *(const uint4*)(Ar1 + k0 + 16 + half * 8);
        // B lane layout: elem i -> K = k0 + (i>>3)*16 + half*8 + (i&7), column = l15
        const uint4* w0 = (const uint4*)(Wr0 + k0 + half * 8);
        b0.q[0] = w0[0];
        b0.q[1] = w0[2];
        const uint4* w1 = (const uint4*)(Wr1 + k0 + half * 8);
        b1.q[0] = w1[0];
        b1.q[1] = w1[2];
        acc00 = WMMA_BF16(a0, b0, acc00);
        acc01 = WMMA_BF16(a0, b1, acc01);
        acc10 = WMMA_BF16(a1, b0, acc10);
        acc11 = WMMA_BF16(a1, b1, acc11);
    }

    const int  col0 = tN + l15;
    const int  col1 = tN + 16 + l15;
    const float bv0 = bias ? bias[col0] : 0.0f;
    const float bv1 = bias ? bias[col1] : 0.0f;
    #pragma unroll
    for (int r = 0; r < 8; ++r) {
        // C layout: VGPR r -> row = r + 8*half, col = l15 (+16/31 split by half)
        int row0 = tM + r + half * 8;
        int row1 = tM + 16 + r + half * 8;
        float x00 = acc00[r] + bv0, x01 = acc01[r] + bv1;
        float x10 = acc10[r] + bv0, x11 = acc11[r] + bv1;
        if (ACT == 1) {
            x00 = 0.5f * x00 * (1.0f + erff(x00 * 0.70710678118654752f));
            x01 = 0.5f * x01 * (1.0f + erff(x01 * 0.70710678118654752f));
            x10 = 0.5f * x10 * (1.0f + erff(x10 * 0.70710678118654752f));
            x11 = 0.5f * x11 * (1.0f + erff(x11 * 0.70710678118654752f));
        }
        size_t i00 = (size_t)row0 * N + col0, i01 = (size_t)row0 * N + col1;
        size_t i10 = (size_t)row1 * N + col0, i11 = (size_t)row1 * N + col1;
        if (WF32)  { Cf[i00] = x00; Cf[i01] = x01; Cf[i10] = x10; Cf[i11] = x11; }
        if (WBF16) { Cb[i00] = f2bf(x00); Cb[i01] = f2bf(x01);
                     Cb[i10] = f2bf(x10); Cb[i11] = f2bf(x11); }
    }
}

// ---------------- host orchestration ----------------
extern "C" void kernel_launch(void* const* d_in, const int* in_sizes, int n_in,
                              void* d_out, int out_size, void* d_ws, size_t ws_size,
                              hipStream_t stream) {
    (void)in_sizes; (void)n_in; (void)out_size; (void)ws_size;
    // setup_inputs() dict insertion order, params flattened in construction order
    const int*   x        = (const int*)d_in[0];
    const float* cantor   = (const float*)d_in[1];
    const int*   routes   = (const int*)d_in[2];
    const float* emb      = (const float*)d_in[3];
    const float* ln_emb_g = (const float*)d_in[4];
    const float* ln_emb_b = (const float*)d_in[5];
    struct LayerP { const float *wq,*wk,*wv,*wo,*bo,*w1,*b1,*w2,*b2,*ln1_g,*ln1_b,*ln2_g,*ln2_b; };
    LayerP L[LAYERS];
    for (int l = 0; l < LAYERS; ++l) {
        const int base = 6 + l * 13;
        L[l].wq    = (const float*)d_in[base + 0];
        L[l].wk    = (const float*)d_in[base + 1];
        L[l].wv    = (const float*)d_in[base + 2];
        L[l].wo    = (const float*)d_in[base + 3];
        L[l].bo    = (const float*)d_in[base + 4];
        L[l].w1    = (const float*)d_in[base + 5];
        L[l].b1    = (const float*)d_in[base + 6];
        L[l].w2    = (const float*)d_in[base + 7];
        L[l].b2    = (const float*)d_in[base + 8];
        L[l].ln1_g = (const float*)d_in[base + 9];
        L[l].ln1_b = (const float*)d_in[base + 10];
        L[l].ln2_g = (const float*)d_in[base + 11];
        L[l].ln2_b = (const float*)d_in[base + 12];
    }
    const float* head_w = (const float*)d_in[6 + LAYERS * 13];
    const float* head_b = (const float*)d_in[7 + LAYERS * 13];

    // workspace carve-up (~74 MB)
    char* ws = (char*)d_ws;
    size_t off = 0;
    auto take = [&](size_t bytes) -> char* {
        char* p = ws + off;
        off = (off + bytes + 255) & ~(size_t)255;
        return p;
    };
    float*          h     = (float*)take((size_t)SEQ * HID * 4);
    unsigned short* hb    = (unsigned short*)take((size_t)SEQ * HID * 2);
    float*          qf    = (float*)take((size_t)SEQ * HID * 4);
    float*          kf    = (float*)take((size_t)SEQ * HID * 4);
    float*          vf    = (float*)take((size_t)SEQ * HID * 4);
    unsigned short* ctxb  = (unsigned short*)take((size_t)SEQ * HID * 2);
    float*          of    = (float*)take((size_t)SEQ * HID * 4);
    unsigned short* ffnb  = (unsigned short*)take((size_t)SEQ * FFN * 2);
    unsigned short *wqb[LAYERS], *wkb[LAYERS], *wvb[LAYERS], *wob[LAYERS],
                   *w1b[LAYERS], *w2b[LAYERS];
    for (int l = 0; l < LAYERS; ++l) {
        wqb[l] = (unsigned short*)take((size_t)HID * HID * 2);
        wkb[l] = (unsigned short*)take((size_t)HID * HID * 2);
        wvb[l] = (unsigned short*)take((size_t)HID * HID * 2);
        wob[l] = (unsigned short*)take((size_t)HID * HID * 2);
        w1b[l] = (unsigned short*)take((size_t)FFN * HID * 2);
        w2b[l] = (unsigned short*)take((size_t)HID * FFN * 2);
    }
    unsigned short* headwb = (unsigned short*)take((size_t)VOCAB * HID * 2);

    auto cvt = [&](const float* in, unsigned short* out, int n) {
        cvt_bf16_kernel<<<dim3((n + 255) / 256), dim3(256), 0, stream>>>(in, out, n);
    };
    for (int l = 0; l < LAYERS; ++l) {
        cvt(L[l].wq, wqb[l], HID * HID);
        cvt(L[l].wk, wkb[l], HID * HID);
        cvt(L[l].wv, wvb[l], HID * HID);
        cvt(L[l].wo, wob[l], HID * HID);
        cvt(L[l].w1, w1b[l], FFN * HID);
        cvt(L[l].w2, w2b[l], HID * FFN);
    }
    cvt(head_w, headwb, VOCAB * HID);

    embed_ln_rope_kernel<<<dim3(SEQ), dim3(HID), 0, stream>>>(
        x, emb, ln_emb_g, ln_emb_b, cantor, h, hb);

    const dim3 blk(256);
    for (int l = 0; l < LAYERS; ++l) {
        // Q, K, V projections (M=8192, N=256, K=256)
        gemm_bf16_wmma<0, 1, 0><<<dim3(HID / 128, SEQ / 64), blk, 0, stream>>>(
            hb, wqb[l], nullptr, qf, nullptr, SEQ, HID, HID);
        gemm_bf16_wmma<0, 1, 0><<<dim3(HID / 128, SEQ / 64), blk, 0, stream>>>(
            hb, wkb[l], nullptr, kf, nullptr, SEQ, HID, HID);
        gemm_bf16_wmma<0, 1, 0><<<dim3(HID / 128, SEQ / 64), blk, 0, stream>>>(
            hb, wvb[l], nullptr, vf, nullptr, SEQ, HID, HID);
        // windowed attention -> bf16 context
        attn_kernel<<<dim3(SEQ, HEADS), dim3(KWIN), 0, stream>>>(qf, kf, vf, routes, ctxb);
        // output projection + bias
        gemm_bf16_wmma<0, 1, 0><<<dim3(HID / 128, SEQ / 64), blk, 0, stream>>>(
            ctxb, wob[l], L[l].bo, of, nullptr, SEQ, HID, HID);
        residual_ln_kernel<<<dim3(SEQ), dim3(HID), 0, stream>>>(
            h, of, L[l].ln1_g, L[l].ln1_b, h, hb);
        // FFN1 + GELU (bf16 out only), FFN2 + bias
        gemm_bf16_wmma<1, 0, 1><<<dim3(FFN / 128, SEQ / 64), blk, 0, stream>>>(
            hb, w1b[l], L[l].b1, nullptr, ffnb, SEQ, FFN, HID);
        gemm_bf16_wmma<0, 1, 0><<<dim3(HID / 128, SEQ / 64), blk, 0, stream>>>(
            ffnb, w2b[l], L[l].b2, of, nullptr, SEQ, HID, FFN);
        residual_ln_kernel<<<dim3(SEQ), dim3(HID), 0, stream>>>(
            h, of, L[l].ln2_g, L[l].ln2_b, h, hb);
    }
    // vocab head: 8192 x 8192 x 256 -> d_out (fp32)
    gemm_bf16_wmma<0, 1, 0><<<dim3(VOCAB / 128, SEQ / 64), blk, 0, stream>>>(
        hb, headwb, head_b, (float*)d_out, nullptr, SEQ, VOCAB, HID);
}
